// TemporalFusionDecoder_4277787427576
// MI455X (gfx1250) — matrix operand order repaired and checked by reading, not compiled, not run on hardware
//
#include <hip/hip_runtime.h>
#include <hip/hip_bf16.h>

// ---------------------------------------------------------------------------
// TemporalFusionDecoder forward for MI455X (gfx1250, wave32, WMMA).
// Output depends only on the last timestep after attention, and V is
// head-shared, so attention collapses to one weighted sum per batch.
// Dominant cost: bf16 WMMA GEMMs over M = B*L = 16384 tokens (~210 GFLOP).
// GEMM: 128x128 block tile, 8 waves (4x2), wave tile 32x64 (2x4 WMMA tiles)
// -> 12 b128 loads per 8 v_wmma (1.5 loads/WMMA) in the steady state.
// Requires ~380 MB of workspace (bump-allocated from d_ws).
// ---------------------------------------------------------------------------

typedef __bf16 bf16;
typedef __attribute__((ext_vector_type(16))) __bf16 v16bf;
typedef __attribute__((ext_vector_type(8)))  __bf16 v8bf;
typedef __attribute__((ext_vector_type(8)))  float  v8f;

// ------------------------------ small utils -------------------------------

__global__ void f32_to_bf16_kernel(const float* __restrict__ in,
                                   bf16* __restrict__ out, long n) {
  long i = (long)blockIdx.x * blockDim.x + threadIdx.x;
  if (i < n) out[i] = (bf16)in[i];
}

// W is K x N (row-major, fp32). Writes Wt as N x K (row-major, bf16).
__global__ void transpose_to_bf16_kernel(const float* __restrict__ W,
                                         bf16* __restrict__ Wt, int K, int N) {
  __shared__ float tile[32][33];
  const int n0 = blockIdx.x * 32, k0 = blockIdx.y * 32;
  const int tx = threadIdx.x, ty = threadIdx.y;      // 32 x 8
#pragma unroll
  for (int i = 0; i < 4; i++) {
    int k = k0 + ty + i * 8, n = n0 + tx;
    if (k < K && n < N) tile[ty + i * 8][tx] = W[(size_t)k * N + n];
  }
  __syncthreads();
#pragma unroll
  for (int i = 0; i < 4; i++) {
    int n = n0 + ty + i * 8, k = k0 + tx;
    if (n < N && k < K) Wt[(size_t)n * K + k] = (bf16)tile[tx][ty + i * 8];
  }
}

// ------------------------------ WMMA GEMM ---------------------------------
// C[M,N] = act( A[M,K](bf16,row stride lda) @ Wt[N,K]^T + bias[N] (+ add) )
// Block tile 128x128, 8 waves (4 rows x 2 cols), wave tile 32x64.
// ACT: 0 = none, 1 = ELU.  OUTB: 0 = f32 out, 1 = bf16 out.
// HASADD: add addv[(row/addDiv)*N + col] (used for the broadcast ctx term).

template <int ACT, int OUTB, int HASADD>
__global__ __launch_bounds__(256) void gemm_wmma(
    const bf16* __restrict__ A, long lda,
    const bf16* __restrict__ Wt,
    const float* __restrict__ bias,
    const float* __restrict__ addv, int addDiv,
    float* __restrict__ Cf, bf16* __restrict__ Cb,
    int M, int N, int K) {
  const int lane = threadIdx.x & 31;
  const int w    = threadIdx.x >> 5;   // 0..7
  const int wm   = w >> 1;             // 0..3
  const int wn   = w & 1;              // 0..1
  const int m0   = blockIdx.y * 128 + wm * 32;
  const int n0   = blockIdx.x * 128 + wn * 64;
  const int lr   = lane & 15;
  const int lh   = lane >> 4;

  // A fragment rows (16x32 tile): lane lr holds row m, khalf lh -> K chunks
  // [kh*8 .. kh*8+7] and [16+kh*8 .. 16+kh*8+7]  (ISA 7.12.2, 16-bit A).
  const bf16* ap[2];
#pragma unroll
  for (int mi = 0; mi < 2; mi++) {
    long r = m0 + mi * 16 + lr;
    if (r > M - 1) r = M - 1;
    ap[mi] = A + r * lda + lh * 8;
  }

  // B fragment (32x16 tile) from Wt (N x K): lane lr holds column n, khalf lh
  // holds K chunks [kh*8 .. kh*8+7] and [16+kh*8 .. 16+kh*8+7] (same as A).
  const bf16* bp[4];
#pragma unroll
  for (int ni = 0; ni < 4; ni++) {
    long c = n0 + ni * 16 + lr;
    if (c > N - 1) c = N - 1;
    bp[ni] = Wt + c * (long)K + lh * 8;
  }

  v8f acc[2][4];
#pragma unroll
  for (int i = 0; i < 2; i++)
#pragma unroll
    for (int j = 0; j < 4; j++) acc[i][j] = (v8f){0, 0, 0, 0, 0, 0, 0, 0};

  for (int k0 = 0; k0 < K; k0 += 32) {
    union { v16bf v; v8bf h[2]; } a[2], b[4];
#pragma unroll
    for (int mi = 0; mi < 2; mi++) {
      a[mi].h[0] = *(const v8bf*)(ap[mi] + k0);
      a[mi].h[1] = *(const v8bf*)(ap[mi] + k0 + 16);
    }
#pragma unroll
    for (int ni = 0; ni < 4; ni++) {
      b[ni].h[0] = *(const v8bf*)(bp[ni] + k0);
      b[ni].h[1] = *(const v8bf*)(bp[ni] + k0 + 16);
    }
#pragma unroll
    for (int mi = 0; mi < 2; mi++)
#pragma unroll
      for (int ni = 0; ni < 4; ni++)
        acc[mi][ni] = __builtin_amdgcn_wmma_f32_16x16x32_bf16(
            false, a[mi].v, false, b[ni].v, (short)0, acc[mi][ni], false, false);
  }

  // D layout: VGPR r -> row (r + 8*lh), column = lr (ISA 7.12.2, 32-bit C/D).
#pragma unroll
  for (int mi = 0; mi < 2; mi++)
#pragma unroll
    for (int ni = 0; ni < 4; ni++) {
      int colg = n0 + ni * 16 + lr;
      if (colg >= N) continue;
      float bv = bias ? bias[colg] : 0.f;
#pragma unroll
      for (int r = 0; r < 8; r++) {
        int rowg = m0 + mi * 16 + lh * 8 + r;
        if (rowg >= M) continue;
        float v = acc[mi][ni][r] + bv;
        if (HASADD) v += addv[(size_t)(rowg / addDiv) * N + colg];
        if (ACT == 1) v = (v > 0.f) ? v : (__expf(v) - 1.f);
        if (OUTB) Cb[(size_t)rowg * N + colg] = (bf16)v;
        else      Cf[(size_t)rowg * N + colg] = v;
      }
    }
}

// -------------------- fused GLU + residual + LayerNorm --------------------
// Y: rows x 2N (a = Y[:, :N], g = Y[:, N:]).  t = a*sigmoid(g) + Res.
// out = LN(t; gamma, beta, eps).  Writes f32 and/or bf16; optionally captures
// rows with (row % Lmod == Lmod-1) into lastF (fp32).   N must be 1024.

__global__ __launch_bounds__(256) void glu_ln_kernel(
    const float* __restrict__ Y,
    const float* __restrict__ Res, long resStride,
    const float* __restrict__ gamma, const float* __restrict__ beta,
    float eps, int N,
    float* __restrict__ outF, bf16* __restrict__ outB,
    float* __restrict__ lastF, int Lmod) {
  const int row = blockIdx.x;
  const int tid = threadIdx.x;
  const float* y   = Y + (size_t)row * 2 * N;
  const float* res = Res + (size_t)row * resStride;

  float t[4];
  float s = 0.f, ss = 0.f;
#pragma unroll
  for (int i = 0; i < 4; i++) {
    int n = tid + i * 256;
    float a = y[n];
    float g = y[N + n];
    float v = a * (1.f / (1.f + __expf(-g))) + res[n];
    t[i] = v;
    s += v;
    ss += v * v;
  }
  __shared__ float rs[256], rss[256];
  rs[tid] = s; rss[tid] = ss;
  __syncthreads();
  for (int k = 128; k > 0; k >>= 1) {
    if (tid < k) { rs[tid] += rs[tid + k]; rss[tid] += rss[tid + k]; }
    __syncthreads();
  }
  float mean = rs[0] / (float)N;
  float var  = rss[0] / (float)N - mean * mean;
  float rstd = rsqrtf(var + eps);
  bool isLast = (lastF != nullptr) && ((row % Lmod) == Lmod - 1);
#pragma unroll
  for (int i = 0; i < 4; i++) {
    int n = tid + i * 256;
    float o = (t[i] - mean) * rstd * gamma[n] + beta[n];
    if (outF) outF[(size_t)row * N + n] = o;
    if (outB) outB[(size_t)row * N + n] = (bf16)o;
    if (isLast) lastF[(size_t)(row / Lmod) * N + n] = o;
  }
}

// ----------------------- last-timestep attention --------------------------
// One block per batch. q only at t = L-1 (causal row admits all 512 keys).
// V is head-shared -> mean over heads of (prob @ v) = (mean_h prob) @ v.
// kv layout: (B*L) x 1088, k cols [0,1024) (head h at h*64), v cols [1024,1088).

__global__ __launch_bounds__(512) void attn_last_kernel(
    const float* __restrict__ qlast,  // B x 1024
    const float* __restrict__ kv,     // (B*512) x 1088
    bf16* __restrict__ mattn) {       // B x 64
  const int b = blockIdx.x;
  const int l = threadIdx.x;          // 0..511
  __shared__ float sc[512][17];
  __shared__ float mx[16], den[16];
  __shared__ float pb[512];
  const float scale = 0.125f;         // 64^-0.5
  const float* q    = qlast + (size_t)b * 1024;
  const float* krow = kv + ((size_t)b * 512 + l) * 1088;
#pragma unroll 2
  for (int h = 0; h < 16; h++) {
    const float* qa = q + h * 64;
    const float* ka = krow + h * 64;
    float d = 0.f;
#pragma unroll 8
    for (int j = 0; j < 64; j++) d += qa[j] * ka[j];
    sc[l][h] = d * scale;
  }
  __syncthreads();
  if (l < 16) {
    float m = -3.4e38f;
    for (int i = 0; i < 512; i++) m = fmaxf(m, sc[i][l]);
    float s = 0.f;
    for (int i = 0; i < 512; i++) s += __expf(sc[i][l] - m);
    mx[l] = m; den[l] = s;
  }
  __syncthreads();
  float p = 0.f;
#pragma unroll 4
  for (int h = 0; h < 16; h++) p += __expf(sc[l][h] - mx[h]) / den[h];
  pb[l] = p * (1.f / 16.f);
  __syncthreads();
  if (l < 64) {
    float acc = 0.f;
    for (int i = 0; i < 512; i++)
      acc += pb[i] * kv[((size_t)b * 512 + i) * 1088 + 1024 + l];
    mattn[(size_t)b * 64 + l] = (bf16)acc;
  }
}

// ------------------------------- launcher ---------------------------------

extern "C" void kernel_launch(void* const* d_in, const int* in_sizes, int n_in,
                              void* d_out, int out_size, void* d_ws,
                              size_t ws_size, hipStream_t stream) {
  const int B = 32, L = 512, H = 1024, CTX = 1024;
  const long M = (long)B * L;  // 16384

  // ---- inputs (setup_inputs dict order, params flattened depth-first) ----
  int ii = 0;
  const float* x  = (const float*)d_in[ii++];  // (B, L, H)
  const float* ce = (const float*)d_in[ii++];  // (B, CTX)
  // enr GRN
  const float* e_skip_w = (const float*)d_in[ii++];
  const float* e_skip_b = (const float*)d_in[ii++];
  const float* e_fc1_w  = (const float*)d_in[ii++];
  const float* e_fc1_b  = (const float*)d_in[ii++];
  const float* e_ctx_w  = (const float*)d_in[ii++];
  const float* e_ctx_b  = (const float*)d_in[ii++];
  const float* e_fc2_w  = (const float*)d_in[ii++];
  const float* e_fc2_b  = (const float*)d_in[ii++];
  const float* e_glu_w  = (const float*)d_in[ii++];
  const float* e_glu_b  = (const float*)d_in[ii++];
  const float* e_ln_g   = (const float*)d_in[ii++];
  const float* e_ln_b   = (const float*)d_in[ii++];
  // pos GRN
  const float* p_skip_w = (const float*)d_in[ii++];
  const float* p_skip_b = (const float*)d_in[ii++];
  const float* p_fc1_w  = (const float*)d_in[ii++];
  const float* p_fc1_b  = (const float*)d_in[ii++];
  const float* p_fc2_w  = (const float*)d_in[ii++];
  const float* p_fc2_b  = (const float*)d_in[ii++];
  const float* p_glu_w  = (const float*)d_in[ii++];
  const float* p_glu_b  = (const float*)d_in[ii++];
  const float* p_ln_g   = (const float*)d_in[ii++];
  const float* p_ln_b   = (const float*)d_in[ii++];
  const float* qkv_w    = (const float*)d_in[ii++];  // (1024, 2112)
  const float* qkv_b    = (const float*)d_in[ii++];
  const float* out_w    = (const float*)d_in[ii++];  // (64, 1024)
  const float* out_b    = (const float*)d_in[ii++];
  const float* attg_w   = (const float*)d_in[ii++];  // (1024, 2048)
  const float* attg_b   = (const float*)d_in[ii++];
  const float* decg_w   = (const float*)d_in[ii++];
  const float* decg_b   = (const float*)d_in[ii++];
  const float* attn_ln_g = (const float*)d_in[ii++];
  const float* attn_ln_b = (const float*)d_in[ii++];
  const float* dec_ln_g  = (const float*)d_in[ii++];
  const float* dec_ln_b  = (const float*)d_in[ii++];

  // ------------------------- workspace layout ---------------------------
  char* base = (char*)d_ws;
  size_t off = 0;
  auto alloc = [&](size_t bytes) -> void* {
    void* p = base + off;
    off = (off + bytes + 255) & ~(size_t)255;
    return p;
  };
  bf16* xb   = (bf16*)alloc(M * H * 2);
  bf16* ceb  = (bf16*)alloc((size_t)B * CTX * 2);
  bf16* wt_eskip = (bf16*)alloc((size_t)H * H * 2);
  bf16* wt_efc1  = (bf16*)alloc((size_t)H * H * 2);
  bf16* wt_ectx  = (bf16*)alloc((size_t)CTX * H * 2);
  bf16* wt_efc2  = (bf16*)alloc((size_t)H * H * 2);
  bf16* wt_eglu  = (bf16*)alloc((size_t)H * 2 * H * 2);
  bf16* wt_pskip = (bf16*)alloc((size_t)H * H * 2);
  bf16* wt_pfc1  = (bf16*)alloc((size_t)H * H * 2);
  bf16* wt_pfc2  = (bf16*)alloc((size_t)H * H * 2);
  bf16* wt_pglu  = (bf16*)alloc((size_t)H * 2 * H * 2);
  bf16* wt_qkv   = (bf16*)alloc((size_t)2112 * H * 2);
  bf16* wt_outw  = (bf16*)alloc((size_t)H * 64 * 2);
  bf16* wt_attg  = (bf16*)alloc((size_t)H * 2 * H * 2);
  bf16* wt_decg  = (bf16*)alloc((size_t)H * 2 * H * 2);
  float* cectx   = (float*)alloc((size_t)B * H * 4);
  float* skipbuf = (float*)alloc(M * H * 4);
  bf16*  hbuf    = (bf16*)alloc(M * H * 2);
  bf16*  h2buf   = (bf16*)alloc(M * H * 2);
  float* ybuf    = (float*)alloc(M * 2 * H * 4);      // also reused for kv
  bf16*  enrb    = (bf16*)alloc(M * H * 2);
  float* enrlast = (float*)alloc((size_t)B * H * 4);
  float* qlast   = (float*)alloc((size_t)B * H * 4);
  bf16*  mattn   = (bf16*)alloc((size_t)B * 64 * 2);
  bf16*  attoutb = (bf16*)alloc((size_t)B * H * 2);
  float* yatt    = (float*)alloc((size_t)B * 2 * H * 4);
  bf16*  xl2b    = (bf16*)alloc((size_t)B * H * 2);
  float* skip2   = (float*)alloc((size_t)B * H * 4);
  bf16*  hhb     = (bf16*)alloc((size_t)B * H * 2);
  bf16*  hh2b    = (bf16*)alloc((size_t)B * H * 2);
  float* y2      = (float*)alloc((size_t)B * 2 * H * 4);
  bf16*  xl3b    = (bf16*)alloc((size_t)B * H * 2);
  float* y3      = (float*)alloc((size_t)B * 2 * H * 4);
  float* kvbuf   = ybuf;  // ybuf dead after the enr glu_ln; kv is 68MB <=128MB

  // --------------------------- conversions ------------------------------
  {
    long n = M * H;
    f32_to_bf16_kernel<<<(unsigned)((n + 255) / 256), 256, 0, stream>>>(x, xb, n);
    long nc = (long)B * CTX;
    f32_to_bf16_kernel<<<(unsigned)((nc + 255) / 256), 256, 0, stream>>>(ce, ceb, nc);
  }
  dim3 tb(32, 8);
  auto tgrid = [](int K, int N) { return dim3((N + 31) / 32, (K + 31) / 32); };
  transpose_to_bf16_kernel<<<tgrid(H, H), tb, 0, stream>>>(e_skip_w, wt_eskip, H, H);
  transpose_to_bf16_kernel<<<tgrid(H, H), tb, 0, stream>>>(e_fc1_w, wt_efc1, H, H);
  transpose_to_bf16_kernel<<<tgrid(CTX, H), tb, 0, stream>>>(e_ctx_w, wt_ectx, CTX, H);
  transpose_to_bf16_kernel<<<tgrid(H, H), tb, 0, stream>>>(e_fc2_w, wt_efc2, H, H);
  transpose_to_bf16_kernel<<<tgrid(H, 2 * H), tb, 0, stream>>>(e_glu_w, wt_eglu, H, 2 * H);
  transpose_to_bf16_kernel<<<tgrid(H, H), tb, 0, stream>>>(p_skip_w, wt_pskip, H, H);
  transpose_to_bf16_kernel<<<tgrid(H, H), tb, 0, stream>>>(p_fc1_w, wt_pfc1, H, H);
  transpose_to_bf16_kernel<<<tgrid(H, H), tb, 0, stream>>>(p_fc2_w, wt_pfc2, H, H);
  transpose_to_bf16_kernel<<<tgrid(H, 2 * H), tb, 0, stream>>>(p_glu_w, wt_pglu, H, 2 * H);
  transpose_to_bf16_kernel<<<tgrid(H, 2112), tb, 0, stream>>>(qkv_w, wt_qkv, H, 2112);
  transpose_to_bf16_kernel<<<tgrid(64, H), tb, 0, stream>>>(out_w, wt_outw, 64, H);
  transpose_to_bf16_kernel<<<tgrid(H, 2 * H), tb, 0, stream>>>(attg_w, wt_attg, H, 2 * H);
  transpose_to_bf16_kernel<<<tgrid(H, 2 * H), tb, 0, stream>>>(decg_w, wt_decg, H, 2 * H);

  auto ggrid = [](long M_, long N_) {
    return dim3((unsigned)((N_ + 127) / 128), (unsigned)((M_ + 127) / 128));
  };
  const dim3 gb(256);

  // ------------------------------ enr GRN --------------------------------
  // cectx[b] = ce[b] @ ctx_w + ctx_b      (B x H)
  gemm_wmma<0, 0, 0><<<ggrid(B, H), gb, 0, stream>>>(
      ceb, (long)CTX, wt_ectx, e_ctx_b, nullptr, 1, cectx, nullptr, B, H, CTX);
  // skip = x @ skip_w + b                 (M x H, f32)
  gemm_wmma<0, 0, 0><<<ggrid(M, H), gb, 0, stream>>>(
      xb, (long)H, wt_eskip, e_skip_b, nullptr, 1, skipbuf, nullptr, (int)M, H, H);
  // h = elu(x @ fc1_w + b + cectx[row/L]) (M x H, bf16)
  gemm_wmma<1, 1, 1><<<ggrid(M, H), gb, 0, stream>>>(
      xb, (long)H, wt_efc1, e_fc1_b, cectx, L, nullptr, hbuf, (int)M, H, H);
  // h2 = h @ fc2_w + b                    (M x H, bf16)
  gemm_wmma<0, 1, 0><<<ggrid(M, H), gb, 0, stream>>>(
      hbuf, (long)H, wt_efc2, e_fc2_b, nullptr, 1, nullptr, h2buf, (int)M, H, H);
  // y = h2 @ glu_w + b                    (M x 2H, f32)
  gemm_wmma<0, 0, 0><<<ggrid(M, 2 * H), gb, 0, stream>>>(
      h2buf, (long)H, wt_eglu, e_glu_b, nullptr, 1, ybuf, nullptr, (int)M, 2 * H, H);
  // enriched = LN(glu(y) + skip); capture last rows fp32
  glu_ln_kernel<<<(unsigned)M, 256, 0, stream>>>(
      ybuf, skipbuf, (long)H, e_ln_g, e_ln_b, 1e-5f, H,
      nullptr, enrb, enrlast, L);

  // --------------------------- K/V and q(last) ---------------------------
  // kv = enriched @ qkv_w[:, 1024:2112] + b   (M x 1088, f32)  [reuses ybuf]
  gemm_wmma<0, 0, 0><<<ggrid(M, 1088), gb, 0, stream>>>(
      enrb, (long)H, wt_qkv + (size_t)1024 * H, qkv_b + 1024, nullptr, 1,
      kvbuf, nullptr, (int)M, 1088, H);
  // q_last = enriched[:, L-1, :] @ qkv_w[:, :1024] + b   (B x 1024, f32)
  gemm_wmma<0, 0, 0><<<ggrid(B, H), gb, 0, stream>>>(
      enrb + (size_t)(L - 1) * H, (long)L * H, wt_qkv, qkv_b, nullptr, 1,
      qlast, nullptr, B, H, H);

  // ------------------------------ attention ------------------------------
  attn_last_kernel<<<B, 512, 0, stream>>>(qlast, kvbuf, mattn);
  // att_out_last = m_attn @ out_w + out_b     (B x H, bf16)
  gemm_wmma<0, 1, 0><<<ggrid(B, H), gb, 0, stream>>>(
      mattn, (long)64, wt_outw, out_b, nullptr, 1, nullptr, attoutb, B, H, 64);

  // ------------------- attn gate + LN (eps = 1e-3) -----------------------
  gemm_wmma<0, 0, 0><<<ggrid(B, 2 * H), gb, 0, stream>>>(
      attoutb, (long)H, wt_attg, attg_b, nullptr, 1, yatt, nullptr, B, 2 * H, H);
  glu_ln_kernel<<<B, 256, 0, stream>>>(
      yatt, enrlast, (long)H, attn_ln_g, attn_ln_b, 1e-3f, H,
      nullptr, xl2b, nullptr, 1 << 30);

  // ------------------------------ pos GRN --------------------------------
  gemm_wmma<0, 0, 0><<<ggrid(B, H), gb, 0, stream>>>(
      xl2b, (long)H, wt_pskip, p_skip_b, nullptr, 1, skip2, nullptr, B, H, H);
  gemm_wmma<1, 1, 0><<<ggrid(B, H), gb, 0, stream>>>(
      xl2b, (long)H, wt_pfc1, p_fc1_b, nullptr, 1, nullptr, hhb, B, H, H);
  gemm_wmma<0, 1, 0><<<ggrid(B, H), gb, 0, stream>>>(
      hhb, (long)H, wt_pfc2, p_fc2_b, nullptr, 1, nullptr, hh2b, B, H, H);
  gemm_wmma<0, 0, 0><<<ggrid(B, 2 * H), gb, 0, stream>>>(
      hh2b, (long)H, wt_pglu, p_glu_b, nullptr, 1, y2, nullptr, B, 2 * H, H);
  glu_ln_kernel<<<B, 256, 0, stream>>>(
      y2, skip2, (long)H, p_ln_g, p_ln_b, 1e-5f, H,
      nullptr, xl3b, nullptr, 1 << 30);

  // -------------------- dec gate + final LN (eps = 1e-3) -----------------
  gemm_wmma<0, 0, 0><<<ggrid(B, 2 * H), gb, 0, stream>>>(
      xl3b, (long)H, wt_decg, decg_b, nullptr, 1, y3, nullptr, B, 2 * H, H);
  // residual = x[:, -1, :] read directly (row stride L*H)
  glu_ln_kernel<<<B, 256, 0, stream>>>(
      y3, x + (size_t)(L - 1) * H, (long)L * H, dec_ln_g, dec_ln_b, 1e-3f, H,
      (float*)d_out, nullptr, nullptr, 1 << 30);

  (void)in_sizes; (void)n_in; (void)out_size; (void)ws_size;
}
